// ReferenceMoEBlock_46420006535171
// MI455X (gfx1250) — matrix operand order repaired and checked by reading, not compiled, not run on hardware
//
#include <hip/hip_runtime.h>
#include <math.h>

typedef __attribute__((ext_vector_type(16))) __bf16 v16bf;
typedef __attribute__((ext_vector_type(2)))  __bf16 v2bf;
typedef __attribute__((ext_vector_type(8)))  float  v8f;
typedef __attribute__((ext_vector_type(2)))  float  v2f;

#define H_DIM 1024
#define I_DIM 512
#define NEXP  16
#define NSLAB 18      // 16 routed + 2 shared-expert halves
#define TOPK  6
#define MB    64      // tokens per block -> 4 WMMA M-tiles, 4x B reuse
#define MTILES 4

// ---- workspace layout (bytes) ----
#define XBF_OFF   (131072L)                         // x_bf[T][1024] bf16
#define GATET_OFF (XBF_OFF + 2L * 2048 * H_DIM)     // 18 x 32nt x 32kb x 512 bf16 tiles
#define WMAT_BYTES (18L * 32 * 32 * 512 * 2)
#define UPT_OFF   (GATET_OFF + WMAT_BYTES)
#define DOWNT_OFF (UPT_OFF + WMAT_BYTES)            // 18 x 64nt x 16kb x 512

// LDS: act[64][512] bf16 + ccol[64] f32 + flag
#define ACT_OFF  0
#define CCOL_OFF 65536
#define FLAG_OFF 65792
#define SMEM_BYTES 65800

struct alignas(16) bf16x8 { __bf16 v[8]; };

__device__ __forceinline__ unsigned pack_bf2(float a, float b) {
  v2f f = {a, b};
  v2bf h = __builtin_convertvector(f, v2bf);   // v_cvt_pk_bf16_f32
  union { v2bf h; unsigned u; } c; c.h = h;
  return c.u;
}
__device__ __forceinline__ __bf16 cvt_bf(float a) {
  v2f f = {a, a};
  v2bf h = __builtin_convertvector(f, v2bf);
  return h.x;
}
// fast silu: bf16-bound output, so v_rcp_f32 accuracy is more than enough
__device__ __forceinline__ float silu_fast(float g) {
  return g * __builtin_amdgcn_rcpf(1.0f + __expf(-g));
}

// A fragment (16x32 bf16 MxK) from row-major base: lane half hh holds
// K = k0+16hh..k0+16hh+15 of row m -> two 16B loads.
__device__ __forceinline__ v16bf load_frag_a(const __bf16* base, int lda, int m, int hh, int k0) {
  const bf16x8* p0 = (const bf16x8*)(base + m * lda + k0 + 16 * hh);
  const bf16x8* p1 = p0 + 1;
  v16bf a;
#pragma unroll
  for (int i = 0; i < 8; ++i) { a[i] = p0->v[i]; a[i + 8] = p1->v[i]; }
  return a;
}

// B fragment from an N-major 16x32 tile (tile[n][k]): lane half hh -> K=16hh..16hh+15,
// contiguous 32B per lane -> two 16B loads (LDS or global).
__device__ __forceinline__ v16bf load_frag_b(const __bf16* tile, int n, int hh) {
  const bf16x8* p0 = (const bf16x8*)(tile + n * 32 + 16 * hh);
  const bf16x8* p1 = p0 + 1;
  v16bf b;
#pragma unroll
  for (int i = 0; i < 8; ++i) { b[i] = p0->v[i]; b[i + 8] = p1->v[i]; }
  return b;
}

// ---------- prep: fp32 weights -> bf16 WMMA-native tiles in workspace ----------
__device__ __forceinline__ void conv_tile(__bf16* dst, const float* __restrict__ src,
                                          int ld, int k0, int n0, int lane) {
  const int kp = lane & 15, nh = lane >> 4;
  const float* r0 = src + (long)(k0 + 2 * kp) * ld + n0 + 8 * nh;
  const float* r1 = r0 + ld;
  float4 a0 = ((const float4*)r0)[0], a1 = ((const float4*)r0)[1];
  float4 b0 = ((const float4*)r1)[0], b1 = ((const float4*)r1)[1];
  float ra[8] = {a0.x, a0.y, a0.z, a0.w, a1.x, a1.y, a1.z, a1.w};
  float rb[8] = {b0.x, b0.y, b0.z, b0.w, b1.x, b1.y, b1.z, b1.w};
#pragma unroll
  for (int j = 0; j < 8; ++j)
    ((unsigned*)(dst + (8 * nh + j) * 32))[kp] = pack_bf2(ra[j], rb[j]);
}

__global__ void __launch_bounds__(256)
prep_w_kernel(const float* __restrict__ w_gate, const float* __restrict__ w_up,
              const float* __restrict__ w_down,
              const float* __restrict__ ws_gate, const float* __restrict__ ws_up,
              const float* __restrict__ ws_down, char* __restrict__ ws) {
  const int lane = threadIdx.x & 31;
  const long gtile = (long)blockIdx.x * 8 + (threadIdx.x >> 5);
  const int kind = (int)(gtile / 18432);       // 0 gate, 1 up, 2 down
  const int t = (int)(gtile % 18432);
  const int e = t / 1024, rem = t % 1024;
  __bf16* dst;
  const float* src; int ld, k0, n0;
  if (kind < 2) {                              // gate/up: [1024][512-slab], 32nt x 32kb
    const int nt = rem / 32, kb = rem % 32;
    const float* base = (kind == 0) ? w_gate : w_up;
    const float* sbase = (kind == 0) ? ws_gate : ws_up;
    if (e < NEXP) { src = base + (long)e * H_DIM * I_DIM; ld = I_DIM; }
    else { src = sbase + (e - NEXP) * I_DIM; ld = 2 * I_DIM; }
    k0 = kb * 32; n0 = nt * 16;
    dst = (__bf16*)(ws + (kind == 0 ? GATET_OFF : UPT_OFF)) + (long)t * 512;
  } else {                                     // down: [512-slab][1024], 64nt x 16kb
    const int nt = rem / 16, kb = rem % 16;
    if (e < NEXP) src = w_down + (long)e * I_DIM * H_DIM;
    else src = ws_down + (long)(e - NEXP) * I_DIM * H_DIM;
    ld = H_DIM; k0 = kb * 32; n0 = nt * 16;
    dst = (__bf16*)(ws + DOWNT_OFF) + (long)t * 512;
  }
  conv_tile(dst, src, ld, k0, n0, lane);
}

__global__ void prep_x_kernel(const float* __restrict__ x, __bf16* __restrict__ xbf) {
  const long i = (long)blockIdx.x * blockDim.x + threadIdx.x;   // float4 index
  float4 f = ((const float4*)x)[i];
  unsigned* d = (unsigned*)xbf;
  d[i * 2]     = pack_bf2(f.x, f.y);
  d[i * 2 + 1] = pack_bf2(f.z, f.w);
}

// =============================== main MoE GEMM ===============================
__global__ void __launch_bounds__(256)
moe_kernel(const __bf16* __restrict__ xbf,
           const __bf16* __restrict__ gateT, const __bf16* __restrict__ upT,
           const __bf16* __restrict__ downT,
           const float* __restrict__ comb, float* __restrict__ out) {
  extern __shared__ char smem[];
  __bf16* act  = (__bf16*)(smem + ACT_OFF);    // [64][512] bf16
  float*  ccol = (float*)(smem + CCOL_OFF);
  int*    flag = (int*)(smem + FLAG_OFF);

  const int tid  = threadIdx.x;
  const int wave = tid >> 5, lane = tid & 31;
  const int hh   = (lane >> 4) & 1, ln = lane & 15;
  const int t0   = blockIdx.x * MB;
  const int slab = blockIdx.y;                 // 0..15 routed, 16..17 shared halves
  const bool routed = slab < NEXP;

  if (tid == 0) *flag = routed ? 0 : 1;
  __syncthreads();
  if (tid < MB) {
    float cw = routed ? comb[(long)(t0 + tid) * NEXP + slab] : 1.0f;
    ccol[tid] = cw;
    if (cw != 0.0f) *flag = 1;
  }
  __syncthreads();
  if (!*flag) return;                          // uniform exit

  const __bf16* xrow = xbf + (long)t0 * H_DIM;

  // ---------------- gate/up GEMMs: K=1024, wave owns 4 N-tiles of I ----------
#pragma unroll 1
  for (int nt = 0; nt < 4; ++nt) {
    const int ntile = wave * 4 + nt;           // 0..31 over I=512
    const __bf16* gT = gateT + ((long)(slab * 32 + ntile) * 32) * 512;
    const __bf16* uT = upT   + ((long)(slab * 32 + ntile) * 32) * 512;
    v8f cg[MTILES], cu[MTILES];
#pragma unroll
    for (int mt = 0; mt < MTILES; ++mt) {
      cg[mt] = (v8f){0.f,0.f,0.f,0.f,0.f,0.f,0.f,0.f};
      cu[mt] = (v8f){0.f,0.f,0.f,0.f,0.f,0.f,0.f,0.f};
    }
#pragma unroll 1
    for (int kb = 0; kb < 32; ++kb) {
      __builtin_prefetch(gT + (kb + 1) * 512, 0, 1);
      __builtin_prefetch(uT + (kb + 1) * 512, 0, 1);
      v16bf bg = load_frag_b(gT + kb * 512, ln, hh);
      v16bf bu = load_frag_b(uT + kb * 512, ln, hh);
      v16bf afr[MTILES];
#pragma unroll
      for (int mt = 0; mt < MTILES; ++mt) {
        afr[mt] = load_frag_a(xrow + mt * 16 * H_DIM, H_DIM, ln, hh, kb * 32);
        cg[mt] = __builtin_amdgcn_wmma_f32_16x16x32_bf16(false, afr[mt], false, bg,
                                                         (short)0, cg[mt], false, false);
      }
#pragma unroll
      for (int mt = 0; mt < MTILES; ++mt)
        cu[mt] = __builtin_amdgcn_wmma_f32_16x16x32_bf16(false, afr[mt], false, bu,
                                                         (short)0, cu[mt], false, false);
    }
    // act[m][col] = comb[m] * silu(g) * u  (C layout: VGPR r <-> M = r + 8*hh)
    const int col0 = ntile * 16;
#pragma unroll
    for (int mt = 0; mt < MTILES; ++mt) {
#pragma unroll
      for (int r = 0; r < 8; ++r) {
        const int m = mt * 16 + r + 8 * hh;
        act[m * I_DIM + col0 + ln] = cvt_bf(silu_fast(cg[mt][r]) * cu[mt][r] * ccol[m]);
      }
    }
  }
  __syncthreads();                             // act complete before down-GEMM

  // ---------------- down GEMM: K=512, wave owns 8 N-tiles of H ---------------
#pragma unroll 1
  for (int nt = 0; nt < 8; ++nt) {
    const int ntile = wave * 8 + nt;           // 0..63 over H=1024
    const __bf16* dT = downT + ((long)(slab * 64 + ntile) * 16) * 512;
    v8f c[MTILES];
#pragma unroll
    for (int mt = 0; mt < MTILES; ++mt)
      c[mt] = (v8f){0.f,0.f,0.f,0.f,0.f,0.f,0.f,0.f};
#pragma unroll 1
    for (int kb = 0; kb < 16; ++kb) {
      __builtin_prefetch(dT + (kb + 1) * 512, 0, 1);
      v16bf b = load_frag_b(dT + kb * 512, ln, hh);
#pragma unroll
      for (int mt = 0; mt < MTILES; ++mt) {
        v16bf a = load_frag_a(act + mt * 16 * I_DIM, I_DIM, ln, hh, kb * 32);
        c[mt] = __builtin_amdgcn_wmma_f32_16x16x32_bf16(false, a, false, b,
                                                        (short)0, c[mt], false, false);
      }
    }
    const int col = ntile * 16 + ln;
#pragma unroll
    for (int mt = 0; mt < MTILES; ++mt)
#pragma unroll
      for (int r = 0; r < 8; ++r)
        atomicAdd(out + (long)(t0 + mt * 16 + r + 8 * hh) * H_DIM + col, c[mt][r]);
  }
}

// ---- router: sigmoid scores, bias-corrected top-6, normalized -> comb[T,16] ----
__global__ void gate_kernel(const float* __restrict__ x, const float* __restrict__ gw,
                            const float* __restrict__ gbias, float* __restrict__ comb, int T) {
  int t = blockIdx.x * blockDim.x + threadIdx.x;
  if (t >= T) return;
  float acc[NEXP];
#pragma unroll
  for (int e = 0; e < NEXP; ++e) acc[e] = 0.0f;
  for (int k = 0; k < H_DIM; k += 4) {
    float4 xv = *(const float4*)(x + (long)t * H_DIM + k);
#pragma unroll
    for (int e = 0; e < NEXP; ++e) {
      const float* g = gw + e * H_DIM + k;
      acc[e] += xv.x * g[0] + xv.y * g[1] + xv.z * g[2] + xv.w * g[3];
    }
  }
  float sc[NEXP]; bool used[NEXP];
#pragma unroll
  for (int e = 0; e < NEXP; ++e) { sc[e] = 1.0f / (1.0f + __expf(-acc[e])); used[e] = false; }
  float wsum = 0.0f;
  for (int kk = 0; kk < TOPK; ++kk) {
    int bi = 0; float bv = -1e30f;
#pragma unroll
    for (int e = 0; e < NEXP; ++e) {
      float v = sc[e] + gbias[e];
      if (!used[e] && v > bv) { bv = v; bi = e; }
    }
    used[bi] = true; wsum += sc[bi];
  }
  const float inv = 1.0f / (wsum + 1e-20f);
#pragma unroll
  for (int e = 0; e < NEXP; ++e) comb[t * NEXP + e] = used[e] ? sc[e] * inv : 0.0f;
}

__global__ void zero_kernel(float4* __restrict__ p) {
  p[(long)blockIdx.x * blockDim.x + threadIdx.x] = (float4){0.f, 0.f, 0.f, 0.f};
}

extern "C" void kernel_launch(void* const* d_in, const int* in_sizes, int n_in,
                              void* d_out, int out_size, void* d_ws, size_t ws_size,
                              hipStream_t stream) {
  const float* x   = (const float*)d_in[0];
  const float* gw  = (const float*)d_in[1];
  const float* gb  = (const float*)d_in[2];
  const float* wg  = (const float*)d_in[3];
  const float* wu  = (const float*)d_in[4];
  const float* wd  = (const float*)d_in[5];
  const float* wsg = (const float*)d_in[6];
  const float* wsu = (const float*)d_in[7];
  const float* wsd = (const float*)d_in[8];
  float* out = (float*)d_out;
  char*  ws  = (char*)d_ws;
  const int T = in_sizes[0] / H_DIM;           // 2048

  float*  comb = (float*)ws;
  __bf16* xbf  = (__bf16*)(ws + XBF_OFF);
  __bf16* gT   = (__bf16*)(ws + GATET_OFF);
  __bf16* uT   = (__bf16*)(ws + UPT_OFF);
  __bf16* dT   = (__bf16*)(ws + DOWNT_OFF);

  (void)hipFuncSetAttribute((const void*)moe_kernel,
                            hipFuncAttributeMaxDynamicSharedMemorySize, SMEM_BYTES);

  zero_kernel<<<(T * H_DIM / 4) / 256, 256, 0, stream>>>((float4*)out);
  gate_kernel<<<(T + 255) / 256, 256, 0, stream>>>(x, gw, gb, comb, T);
  prep_x_kernel<<<(T * H_DIM / 4) / 256, 256, 0, stream>>>(x, xbf);
  prep_w_kernel<<<(3 * 18432) / 8, 256, 0, stream>>>(wg, wu, wd, wsg, wsu, wsd, ws);
  dim3 grid(T / MB, NSLAB);
  moe_kernel<<<grid, 256, SMEM_BYTES, stream>>>(xbf, gT, uT, dT, comb, out);
}
